// Attention_4810363372413
// MI455X (gfx1250) — matrix operand order repaired and checked by reading, not compiled, not run on hardware
//
#include <hip/hip_runtime.h>

// ---------------------------------------------------------------------------
// MI455X / gfx1250 attention layer.
// All matmuls via v_wmma_f32_16x16x32_bf16 (f32 accumulate).
// Problem constants from the reference: B=2, S=2048, E=2048, HQ=32, HK=8, D=64.
// Weights are converted f32->bf16 ONCE and stored pre-transposed [N][K] so the
// GEMM hot loops use only b128 global loads + b128 LDS stores (no per-iteration
// scalar transpose).
// ---------------------------------------------------------------------------

typedef __attribute__((ext_vector_type(16))) __bf16 bf16x16;
typedef __attribute__((ext_vector_type(8)))  float  floatx8;

#define S_LEN 2048
#define E_DIM 2048
#define HQ_N  32
#define HK_N  8
#define HD    64
#define B_N   2
#define M_TOT (B_N * S_LEN)               // 4096 rows of x / o
#define QKV_N (HQ_N * HD + 2 * HK_N * HD) // 3072 fused projection cols

union ABFrag { uint4 u[2]; bf16x16 v; };

__device__ __forceinline__ unsigned short f32_to_bf16(float x) {
  unsigned int u = __float_as_uint(x);
  u += 0x7FFFu + ((u >> 16) & 1u);   // round-to-nearest-even
  return (unsigned short)(u >> 16);
}

// A-matrix fragment, 16x32 bf16, per ISA layout:
// lane L holds row M=L&15; K chunks [kh..kh+7] and [kh+16..kh+23], kh = (L<16)?0:8.
__device__ __forceinline__ bf16x16 load_frag_a(const unsigned short* base, int ld) {
  const int lane = threadIdx.x & 31;
  const unsigned short* p = base + (lane & 15) * ld + ((lane & 16) ? 8 : 0);
  ABFrag f;
  f.u[0] = *reinterpret_cast<const uint4*>(p);
  f.u[1] = *reinterpret_cast<const uint4*>(p + 16);
  return f.v;
}

// B-matrix fragment, 32x16 bf16, read from [N][K] (transposed) storage:
// lane L holds column N=L&15; K chunks [kh..kh+7] and [kh+16..kh+23], kh = (L<16)?0:8.
__device__ __forceinline__ bf16x16 load_frag_b(const unsigned short* base, int ld) {
  const int lane = threadIdx.x & 31;
  const unsigned short* p = base + (lane & 15) * ld + ((lane & 16) ? 8 : 0);
  ABFrag f;
  f.u[0] = *reinterpret_cast<const uint4*>(p);
  f.u[1] = *reinterpret_cast<const uint4*>(p + 16);
  return f.v;
}

__device__ __forceinline__ floatx8 wmma_bf16(bf16x16 a, bf16x16 b, floatx8 c) {
  return __builtin_amdgcn_wmma_f32_16x16x32_bf16(false, a, false, b, (short)0, c,
                                                 false, false);
}

// ---------------------------------------------------------------------------
// f32 -> bf16 conversion kernels
// ---------------------------------------------------------------------------
__global__ void k_cvt(const float* __restrict__ in, unsigned short* __restrict__ out,
                      int n) {
  int i = blockIdx.x * 256 + threadIdx.x;
  if (i < n) out[i] = f32_to_bf16(in[i]);
}

// Transposing placement: input [K=2048][cols] f32 -> output rows [coloff+c][K]
// bf16 (row length E_DIM). One-time cost; removes transpose from GEMM hot loops.
__global__ void k_cvt_place_t(const float* __restrict__ in,
                              unsigned short* __restrict__ out,
                              int cols, int coloff, int n) {
  int i = blockIdx.x * 256 + threadIdx.x;
  if (i < n) {
    int r = i / cols, c = i - r * cols;          // r = K index, c = N index
    out[(size_t)(coloff + c) * E_DIM + r] = f32_to_bf16(in[i]);
  }
}

// ---------------------------------------------------------------------------
// Kernel 2: fused QKV projection + RoPE.
// C[4096 x 3072] = X[4096 x 2048] * W[2048 x 3072]; tile 128x128x64; 8 waves,
// each wave 32x64 (one full head wide -> RoPE pairs (d, d+32) live in frags
// nb and nb+2 of the same wave). W supplied transposed [3072][2048].
// ---------------------------------------------------------------------------
#define GBM 128
#define GBN 128
#define GBK 64
#define GLD 72   // padded LDS row (bf16 elems)

__global__ __launch_bounds__(256) void k_qkv_rope(
    const unsigned short* __restrict__ X,   // [4096][2048] bf16
    const unsigned short* __restrict__ WT,  // [3072][2048] bf16 (transposed Q|K|V)
    const float* __restrict__ rc,           // [2048][32]
    const float* __restrict__ rsn,          // [2048][32]
    unsigned short* __restrict__ Q,         // [B][HQ][S][D]
    unsigned short* __restrict__ K,         // [B][HK][S][D]
    unsigned short* __restrict__ V) {       // [B][HK][D][S] (transposed)
  __shared__ __align__(16) unsigned short sA[GBM * GLD];
  __shared__ __align__(16) unsigned short sB[GBN * GLD];
  const int tid = threadIdx.x;
  const int lane = tid & 31;
  const int wave = tid >> 5;
  const int wm = wave & 3;        // 4 waves along M
  const int wn = wave >> 2;       // 2 waves along N
  const int m0 = blockIdx.y * GBM;
  const int n0 = blockIdx.x * GBN;

  const floatx8 z8 = {0.f, 0.f, 0.f, 0.f, 0.f, 0.f, 0.f, 0.f};
  floatx8 acc[2][4];
#pragma unroll
  for (int i = 0; i < 2; ++i)
#pragma unroll
    for (int j = 0; j < 4; ++j) acc[i][j] = z8;

  const int tr = tid >> 1, tk = (tid & 1) * 32;   // 128 rows x 64 k, 32 elems/thread

  for (int kt = 0; kt < E_DIM; kt += GBK) {
    {
      const uint4* g =
          reinterpret_cast<const uint4*>(X + (size_t)(m0 + tr) * E_DIM + kt + tk);
      uint4 a0 = g[0], a1 = g[1], a2 = g[2], a3 = g[3];
      uint4* d = reinterpret_cast<uint4*>(&sA[tr * GLD + tk]);
      d[0] = a0; d[1] = a1; d[2] = a2; d[3] = a3;
    }
    {
      const uint4* g =
          reinterpret_cast<const uint4*>(WT + (size_t)(n0 + tr) * E_DIM + kt + tk);
      uint4 b0 = g[0], b1 = g[1], b2 = g[2], b3 = g[3];
      uint4* d = reinterpret_cast<uint4*>(&sB[tr * GLD + tk]);
      d[0] = b0; d[1] = b1; d[2] = b2; d[3] = b3;
    }
    __syncthreads();
#pragma unroll
    for (int ks = 0; ks < GBK; ks += 32) {
      bf16x16 af0 = load_frag_a(&sA[(wm * 32) * GLD + ks], GLD);
      bf16x16 af1 = load_frag_a(&sA[(wm * 32 + 16) * GLD + ks], GLD);
#pragma unroll
      for (int nb = 0; nb < 4; ++nb) {
        bf16x16 bfr = load_frag_b(&sB[(wn * 64 + nb * 16) * GLD + ks], GLD);
        acc[0][nb] = wmma_bf16(af0, bfr, acc[0][nb]);
        acc[1][nb] = wmma_bf16(af1, bfr, acc[1][nb]);
      }
    }
    __syncthreads();
  }

  // Epilogue: RoPE (Q and K regions) + scatter to Q/K/V buffers.
  const int half = (lane >> 4) & 1;
  const int nn = lane & 15;
  const int col0 = n0 + wn * 64;     // head-aligned (multiple of 64)
  const int mrow0 = m0 + wm * 32;
  const bool isV = (col0 >= 2560);

  if (!isV) {
#pragma unroll
    for (int mi = 0; mi < 2; ++mi)
#pragma unroll
      for (int nb = 0; nb < 2; ++nb)
#pragma unroll
        for (int j = 0; j < 8; ++j) {
          int m = mrow0 + mi * 16 + half * 8 + j;
          int s = m & (S_LEN - 1);
          int d = nb * 16 + nn;                 // 0..31
          float c = rc[s * 32 + d];
          float sn = rsn[s * 32 + d];
          float t1 = acc[mi][nb][j];
          float t2 = acc[mi][nb + 2][j];
          acc[mi][nb][j]     = t1 * c - t2 * sn;
          acc[mi][nb + 2][j] = t2 * c + t1 * sn;
        }
  }

#pragma unroll
  for (int mi = 0; mi < 2; ++mi)
#pragma unroll
    for (int nb = 0; nb < 4; ++nb)
#pragma unroll
      for (int j = 0; j < 8; ++j) {
        int m = mrow0 + mi * 16 + half * 8 + j;
        int batch = m >> 11;
        int s = m & (S_LEN - 1);
        int d = nb * 16 + nn;                   // 0..63 within head
        unsigned short val = f32_to_bf16(acc[mi][nb][j]);
        if (col0 < 2048) {
          int h = col0 >> 6;
          Q[(((size_t)batch * HQ_N + h) * S_LEN + s) * HD + d] = val;
        } else if (col0 < 2560) {
          int h = (col0 - 2048) >> 6;
          K[(((size_t)batch * HK_N + h) * S_LEN + s) * HD + d] = val;
        } else {
          int h = (col0 - 2560) >> 6;
          V[(((size_t)batch * HK_N + h) * HD + d) * S_LEN + s] = val;
        }
      }
}

// ---------------------------------------------------------------------------
// Kernel 3: flash attention. Block = 8 waves x 16 queries = 128 queries of one
// (b, hq). K/V (shared kv head, GQA hq>>2) streamed in 64-key LDS tiles.
// Scores and PV are bf16 WMMA; softmax in f32 with half-wave shfl reductions.
// ---------------------------------------------------------------------------
#define ALD 72   // padded LDS row (bf16 elems)

__global__ __launch_bounds__(256) void k_attn(
    const unsigned short* __restrict__ Qb,  // [B][HQ][S][D]
    const unsigned short* __restrict__ Kb,  // [B][HK][S][D]
    const unsigned short* __restrict__ Vb,  // [B][HK][D][S]
    const float* __restrict__ mask,         // [S][S]
    unsigned short* __restrict__ Ob) {      // [B][S][HQ*D]
  __shared__ __align__(16) unsigned short sK[64 * ALD];
  __shared__ __align__(16) unsigned short sV[64 * ALD];
  __shared__ __align__(16) unsigned short sP[8 * 16 * ALD];

  const int tid = threadIdx.x;
  const int lane = tid & 31;
  const int wave = tid >> 5;
  const int half = (lane >> 4) & 1;
  const int nn = lane & 15;

  const int bh = blockIdx.y;                 // [0, B*HQ)
  const int b = bh >> 5, hq = bh & 31, hk = hq >> 2;
  const int q0 = blockIdx.x * 128 + wave * 16;

  const unsigned short* qptr =
      Qb + (((size_t)b * HQ_N + hq) * S_LEN + q0) * HD;
  bf16x16 qf0 = load_frag_a(qptr, HD);        // d 0..31
  bf16x16 qf1 = load_frag_a(qptr + 32, HD);   // d 32..63

  const unsigned short* kbase = Kb + ((size_t)b * HK_N + hk) * S_LEN * HD;
  const unsigned short* vbase = Vb + ((size_t)b * HK_N + hk) * HD * S_LEN;

  const floatx8 z8 = {0.f, 0.f, 0.f, 0.f, 0.f, 0.f, 0.f, 0.f};
  floatx8 oacc[4];
#pragma unroll
  for (int f = 0; f < 4; ++f) oacc[f] = z8;
  float mrun[8], lrun[8];
#pragma unroll
  for (int j = 0; j < 8; ++j) { mrun[j] = -1.0e30f; lrun[j] = 0.f; }

  const int lr = tid >> 2, lseg = (tid & 3) * 16;   // tile loaders: 64x64

  for (int kb = 0; kb < S_LEN; kb += 64) {
    __syncthreads();   // prior iteration done reading sK/sV
    {
      const uint4* g =
          reinterpret_cast<const uint4*>(kbase + (size_t)(kb + lr) * HD + lseg);
      uint4 a0 = g[0], a1 = g[1];
      *reinterpret_cast<uint4*>(&sK[lr * ALD + lseg]) = a0;
      *reinterpret_cast<uint4*>(&sK[lr * ALD + lseg + 8]) = a1;
    }
    {
      const uint4* g =
          reinterpret_cast<const uint4*>(vbase + (size_t)lr * S_LEN + kb + lseg);
      uint4 a0 = g[0], a1 = g[1];
      *reinterpret_cast<uint4*>(&sV[lr * ALD + lseg]) = a0;
      *reinterpret_cast<uint4*>(&sV[lr * ALD + lseg + 8]) = a1;
    }
    __syncthreads();

    // scores: 16 queries x 64 keys
    floatx8 sc[4];
#pragma unroll
    for (int nb = 0; nb < 4; ++nb) {
      sc[nb] = z8;
      bf16x16 b0 = load_frag_b(&sK[nb * 16 * ALD], ALD);       // d 0..31
      bf16x16 b1 = load_frag_b(&sK[nb * 16 * ALD + 32], ALD);  // d 32..63
      sc[nb] = wmma_bf16(qf0, b0, sc[nb]);
      sc[nb] = wmma_bf16(qf1, b1, sc[nb]);
    }
    // + attn_mask (zeros in this harness, kept for fidelity; no 1/sqrt(d))
#pragma unroll
    for (int nb = 0; nb < 4; ++nb)
#pragma unroll
      for (int j = 0; j < 8; ++j)
        sc[nb][j] += mask[(size_t)(q0 + half * 8 + j) * S_LEN + kb + nb * 16 + nn];

    // online softmax; row = (half, j) spread over 16 lanes of this half
#pragma unroll
    for (int j = 0; j < 8; ++j) {
      float lm = fmaxf(fmaxf(sc[0][j], sc[1][j]), fmaxf(sc[2][j], sc[3][j]));
#pragma unroll
      for (int o = 1; o < 16; o <<= 1) lm = fmaxf(lm, __shfl_xor(lm, o, 32));
      float mnew = fmaxf(mrun[j], lm);
      float scale = __expf(mrun[j] - mnew);
      float rsum = 0.f;
#pragma unroll
      for (int nb = 0; nb < 4; ++nb) {
        float p = __expf(sc[nb][j] - mnew);
        sc[nb][j] = p;
        rsum += p;
      }
#pragma unroll
      for (int o = 1; o < 16; o <<= 1) rsum += __shfl_xor(rsum, o, 32);
      lrun[j] = lrun[j] * scale + rsum;
      mrun[j] = mnew;
#pragma unroll
      for (int f = 0; f < 4; ++f) oacc[f][j] *= scale;
    }

    // P (C-layout) -> per-wave LDS scratch (A-layout source)
    unsigned short* pbase = &sP[wave * 16 * ALD];
#pragma unroll
    for (int nb = 0; nb < 4; ++nb)
#pragma unroll
      for (int j = 0; j < 8; ++j)
        pbase[(half * 8 + j) * ALD + nb * 16 + nn] = f32_to_bf16(sc[nb][j]);

    // O += P * V
#pragma unroll
    for (int kk = 0; kk < 2; ++kk) {
      bf16x16 pf = load_frag_a(pbase + kk * 32, ALD);
#pragma unroll
      for (int nb = 0; nb < 4; ++nb) {
        bf16x16 vf = load_frag_b(&sV[nb * 16 * ALD + kk * 32], ALD);
        oacc[nb] = wmma_bf16(pf, vf, oacc[nb]);
      }
    }
  }

  // finalize: O /= l, write [b][s][hq*64+d] bf16
#pragma unroll
  for (int j = 0; j < 8; ++j) {
    float inv = 1.f / lrun[j];
    int s = q0 + half * 8 + j;
#pragma unroll
    for (int nb = 0; nb < 4; ++nb) {
      int d = nb * 16 + nn;
      Ob[((size_t)b * S_LEN + s) * E_DIM + hq * HD + d] =
          f32_to_bf16(oacc[nb][j] * inv);
    }
  }
}

// ---------------------------------------------------------------------------
// Kernel 4: output projection. out[4096 x 2048] = O[4096 x 2048] * Wo[2048 x 2048]
// Same tiling as k_qkv_rope; Wo supplied transposed [E][HQ*D]; f32 epilogue.
// ---------------------------------------------------------------------------
__global__ __launch_bounds__(256) void k_out_gemm(
    const unsigned short* __restrict__ A,    // [4096][2048] bf16
    const unsigned short* __restrict__ BwT,  // [2048][2048] bf16 (transposed Wo)
    float* __restrict__ out) {               // [4096][2048] f32
  __shared__ __align__(16) unsigned short sA[GBM * GLD];
  __shared__ __align__(16) unsigned short sB[GBN * GLD];
  const int tid = threadIdx.x;
  const int lane = tid & 31;
  const int wave = tid >> 5;
  const int wm = wave & 3;
  const int wn = wave >> 2;
  const int m0 = blockIdx.y * GBM;
  const int n0 = blockIdx.x * GBN;

  const floatx8 z8 = {0.f, 0.f, 0.f, 0.f, 0.f, 0.f, 0.f, 0.f};
  floatx8 acc[2][4];
#pragma unroll
  for (int i = 0; i < 2; ++i)
#pragma unroll
    for (int j = 0; j < 4; ++j) acc[i][j] = z8;

  const int tr = tid >> 1, tk = (tid & 1) * 32;

  for (int kt = 0; kt < E_DIM; kt += GBK) {
    {
      const uint4* g =
          reinterpret_cast<const uint4*>(A + (size_t)(m0 + tr) * E_DIM + kt + tk);
      uint4 a0 = g[0], a1 = g[1], a2 = g[2], a3 = g[3];
      uint4* d = reinterpret_cast<uint4*>(&sA[tr * GLD + tk]);
      d[0] = a0; d[1] = a1; d[2] = a2; d[3] = a3;
    }
    {
      const uint4* g =
          reinterpret_cast<const uint4*>(BwT + (size_t)(n0 + tr) * E_DIM + kt + tk);
      uint4 b0 = g[0], b1 = g[1], b2 = g[2], b3 = g[3];
      uint4* d = reinterpret_cast<uint4*>(&sB[tr * GLD + tk]);
      d[0] = b0; d[1] = b1; d[2] = b2; d[3] = b3;
    }
    __syncthreads();
#pragma unroll
    for (int ks = 0; ks < GBK; ks += 32) {
      bf16x16 af0 = load_frag_a(&sA[(wm * 32) * GLD + ks], GLD);
      bf16x16 af1 = load_frag_a(&sA[(wm * 32 + 16) * GLD + ks], GLD);
#pragma unroll
      for (int nb = 0; nb < 4; ++nb) {
        bf16x16 bfr = load_frag_b(&sB[(wn * 64 + nb * 16) * GLD + ks], GLD);
        acc[0][nb] = wmma_bf16(af0, bfr, acc[0][nb]);
        acc[1][nb] = wmma_bf16(af1, bfr, acc[1][nb]);
      }
    }
    __syncthreads();
  }

  const int half = (lane >> 4) & 1;
  const int nn = lane & 15;
  const int col0 = n0 + wn * 64;
  const int mrow0 = m0 + wm * 32;
#pragma unroll
  for (int mi = 0; mi < 2; ++mi)
#pragma unroll
    for (int nb = 0; nb < 4; ++nb)
#pragma unroll
      for (int j = 0; j < 8; ++j) {
        int m = mrow0 + mi * 16 + half * 8 + j;
        out[(size_t)m * E_DIM + col0 + nb * 16 + nn] = acc[mi][nb][j];
      }
}

// ---------------------------------------------------------------------------
// Launcher. Workspace layout (bytes):
//   xb   @ 0         16,777,216   x as bf16                 [4096][2048]
//   WcT  @ 16777216  12,582,912   (Wq|Wk|Wv)^T as bf16      [3072][2048]
//   qb   @ 29360128  16,777,216   q bf16 (post-RoPE)        [B][HQ][S][D]
//   kb   @ 46137344   4,194,304   k bf16 (post-RoPE)        [B][HK][S][D]
//   vb   @ 50331648   4,194,304   v bf16 (transposed)       [B][HK][D][S]
//   ob   @ 54525952  16,777,216   attn output bf16          [B][S][HQ*D]
//   wobT @ 71303168   8,388,608   Wo^T bf16                 [2048][2048]
// total ~76 MB.
// ---------------------------------------------------------------------------
extern "C" void kernel_launch(void* const* d_in, const int* in_sizes, int n_in,
                              void* d_out, int out_size, void* d_ws, size_t ws_size,
                              hipStream_t stream) {
  (void)in_sizes; (void)n_in; (void)out_size; (void)ws_size;
  const float* x    = (const float*)d_in[0];
  const float* rc   = (const float*)d_in[1];
  const float* rsn  = (const float*)d_in[2];
  const float* mask = (const float*)d_in[3];
  const float* Wq   = (const float*)d_in[4];
  const float* Wk   = (const float*)d_in[5];
  const float* Wv   = (const float*)d_in[6];
  const float* Wo   = (const float*)d_in[7];
  float* out = (float*)d_out;

  char* ws = (char*)d_ws;
  unsigned short* xb   = (unsigned short*)(ws + 0);
  unsigned short* WcT  = (unsigned short*)(ws + 16777216);
  unsigned short* qb   = (unsigned short*)(ws + 29360128);
  unsigned short* kb   = (unsigned short*)(ws + 46137344);
  unsigned short* vb   = (unsigned short*)(ws + 50331648);
  unsigned short* ob   = (unsigned short*)(ws + 54525952);
  unsigned short* wobT = (unsigned short*)(ws + 71303168);

  // Phase 1: f32 -> bf16 (weights transposed to [N][K] once, here)
  {
    int n = M_TOT * E_DIM;                       // 8,388,608
    k_cvt<<<(n + 255) / 256, 256, 0, stream>>>(x, xb, n);
  }
  {
    int n = E_DIM * HQ_N * HD;                   // 4,194,304
    k_cvt_place_t<<<(n + 255) / 256, 256, 0, stream>>>(Wq, WcT, HQ_N * HD, 0, n);
  }
  {
    int n = E_DIM * HK_N * HD;                   // 1,048,576
    k_cvt_place_t<<<(n + 255) / 256, 256, 0, stream>>>(Wk, WcT, HK_N * HD, 2048, n);
    k_cvt_place_t<<<(n + 255) / 256, 256, 0, stream>>>(Wv, WcT, HK_N * HD, 2560, n);
  }
  {
    int n = HQ_N * HD * E_DIM;                   // 4,194,304  ([h*D+d][e] -> [e][h*D+d])
    k_cvt_place_t<<<(n + 255) / 256, 256, 0, stream>>>(Wo, wobT, E_DIM, 0, n);
  }

  // Phase 2: fused QKV GEMM + RoPE
  k_qkv_rope<<<dim3(QKV_N / GBN, M_TOT / GBM), 256, 0, stream>>>(
      xb, WcT, rc, rsn, qb, kb, vb);

  // Phase 3: flash attention (GQA)
  k_attn<<<dim3(S_LEN / 128, B_N * HQ_N), 256, 0, stream>>>(qb, kb, vb, mask, ob);

  // Phase 4: output projection -> f32 d_out
  k_out_gemm<<<dim3(E_DIM / GBN, M_TOT / GBM), 256, 0, stream>>>(ob, wobT, out);
}
